// Wcompute_33552284517032
// MI455X (gfx1250) — matrix operand order repaired and checked by reading, not compiled, not run on hardware
//
#include <hip/hip_runtime.h>

// Problem constants (match reference)
#define BB 4
#define NN 512
#define CC 128
#define NFE 96
#define HH 192

typedef __attribute__((ext_vector_type(16))) _Float16 v16h;
typedef __attribute__((ext_vector_type(8)))  _Float16 v8h;
typedef __attribute__((ext_vector_type(8)))  float    v8f;
typedef __attribute__((ext_vector_type(4)))  float    v4f;

// Pre-swizzled weight-fragment layout in workspace (element offsets, f16):
// one 16x16(xK=32) B tile = 32 lanes x 16 f16 = 512 elements.
#define KT1 4
#define NT1 12
#define KT2 6
#define NT2 12
#define KT3 6
#define NT3 6
#define KT4 3
#define NT4 6
#define KT5 3
#define OFF_W1 0
#define OFF_W2 (OFF_W1 + KT1*NT1*512)   // 24576
#define OFF_W3 (OFF_W2 + KT2*NT2*512)   // 61440
#define OFF_W4 (OFF_W3 + KT3*NT3*512)   // 79872
#define OFF_W5 (OFF_W4 + KT4*NT4*512)   // 89088
#define WS_F16_TOTAL (OFF_W5 + KT5*512) // 90624 f16 = ~181 KB (fits d_ws)

// ---------------------------------------------------------------------------
// Pre-swizzle f32 weights [Nout, K] into per-lane WMMA B fragments (f16).
// B-matrix 16-bit 32x16 striping (CDNA5 ISA 7.12.2 / sparse-B table):
//   lane l -> column n = (l & 15); element e -> K = (e>>3)*16 + (l>>4)*8 + (e&7)
// ---------------------------------------------------------------------------
__global__ void prep_wfrag(const float* __restrict__ w, _Float16* __restrict__ dst,
                           int K, int nNT, int total) {
    int idx = blockIdx.x * blockDim.x + threadIdx.x;
    if (idx >= total) return;
    int e    = idx & 15;
    int lane = (idx >> 4) & 31;
    int tile = idx >> 9;
    int nt = tile % nNT;
    int kt = tile / nNT;
    int n = nt * 16 + (lane & 15);
    int k = kt * 32 + ((e >> 3) << 4) + ((lane >> 4) << 3) + (e & 7);
    dst[idx] = (_Float16)w[n * K + k];
}

// w5 is [1, 96]: broadcast w5[k] into every column of the B fragment so any
// column of the WMMA result holds the logit for its row.
__global__ void prep_w5frag(const float* __restrict__ w, _Float16* __restrict__ dst,
                            int total) {
    int idx = blockIdx.x * blockDim.x + threadIdx.x;
    if (idx >= total) return;
    int e    = idx & 15;
    int lane = (idx >> 4) & 31;
    int kt   = idx >> 9;
    int k = kt * 32 + ((e >> 3) << 4) + ((lane >> 4) << 3) + (e & 7);
    dst[idx] = (_Float16)w[k];
}

// ---------------------------------------------------------------------------
// Fragment loaders
// ---------------------------------------------------------------------------
__device__ __forceinline__ v16h ld_bfrag(const _Float16* __restrict__ wf, int tile, int lane) {
    // 32 B per lane, 32 B aligned -> two global_load_b128
    return *(const v16h*)(wf + tile * 512 + lane * 16);
}

// A-matrix 16-bit 16x32 striping: lane l -> row m=(l&15);
// element e -> K = (e>>3)*16 + ((l>>4)*8) + (e&7)  (two contiguous runs of 8)
__device__ __forceinline__ v16h ld_afrag(const _Float16* __restrict__ buf,
                                         int lane, int kf, int stride) {
    const int r  = lane & 15;
    const int kb = ((lane >> 4) << 3) + (kf << 5);
    const _Float16* p = buf + r * stride + kb;
    v8h lo = *(const v8h*)(p);        // K = kb .. kb+7       (ds_load_b128)
    v8h hc = *(const v8h*)(p + 16);   // K = kb+16 .. kb+23   (ds_load_b128)
    v16h a;
#pragma unroll
    for (int e = 0; e < 8; ++e) { a[e] = lo[e]; a[e + 8] = hc[e]; }
    return a;
}

// GEMM tile row: acc[16x(NT*16)] = A[16xK] * B[Kx(NT*16)], +bias, leaky-relu,
// f16 store to LDS in C/D layout (lane -> col n=(l&15), rows v+8*(l>>4)).
// Epilogue uses packed f32 math + v_cvt_pk_rtz_f16_f32 per row pair.
template <int KF, int NT>
__device__ __forceinline__ void gemm_store(const v16h* a,
                                           const _Float16* __restrict__ wf,
                                           const float* __restrict__ bias,
                                           _Float16* __restrict__ outbuf,
                                           int ostride, int lane) {
    const int hi = lane >> 4;
    const int nl = lane & 15;
#pragma unroll
    for (int nt = 0; nt < NT; ++nt) {
        v8f acc = {};
#pragma unroll
        for (int kf = 0; kf < KF; ++kf) {
            v16h bfr = ld_bfrag(wf, kf * NT + nt, lane);
            acc = __builtin_amdgcn_wmma_f32_16x16x32_f16(
                false, a[kf], false, bfr, (short)0, acc, false, false);
        }
        const float bn = bias[nt * 16 + nl];
        v8f t = acc + bn;      // packed f32 adds (vector-scalar splat)
        v8f u = t * 0.01f;     // packed f32 muls
        _Float16* q = outbuf + (hi << 3) * ostride + nt * 16 + nl;
#pragma unroll
        for (int p = 0; p < 4; ++p) {
            float e0 = fmaxf(t[2 * p],     u[2 * p]);      // leaky-relu
            float e1 = fmaxf(t[2 * p + 1], u[2 * p + 1]);
            auto pk = __builtin_amdgcn_cvt_pkrtz(e0, e1);  // v_cvt_pk_rtz_f16_f32
            q[(2 * p) * ostride]     = (_Float16)pk[0];    // ds_store_b16
            q[(2 * p + 1) * ostride] = (_Float16)pk[1];    // ds_store_b16_d16_hi
        }
    }
}

template <int KF, int NT>
__device__ __forceinline__ void run_layer(const _Float16* __restrict__ inbuf, int istride,
                                          const _Float16* __restrict__ wf,
                                          const float* __restrict__ bias,
                                          _Float16* __restrict__ outbuf, int ostride,
                                          int lane) {
    v16h a[KF];
#pragma unroll
    for (int kf = 0; kf < KF; ++kf) a[kf] = ld_afrag(inbuf, lane, kf, istride);
    gemm_store<KF, NT>(a, wf, bias, outbuf, ostride, lane);
}

// ---------------------------------------------------------------------------
// Fused edge-MLP + masked softmax. One workgroup per (b, i); 4 waves, each
// wave owns a 16-row j-tile and loops 8x to cover j = 0..511.
// ---------------------------------------------------------------------------
__global__ __launch_bounds__(128) void edge_mlp_softmax(
    const float* __restrict__ x,
    const float* __restrict__ b1, const float* __restrict__ b2,
    const float* __restrict__ b3, const float* __restrict__ b4,
    const float* __restrict__ b5,
    const _Float16* __restrict__ wf,
    float* __restrict__ out) {
    __shared__ __align__(32) _Float16 s_buf[4][2][16 * HH];  // 48 KB: per-wave ping-pong
    __shared__ __align__(16) float s_xi[CC];
    __shared__ __align__(16) float s_wrow[NN];
    __shared__ float s_red[128];
    __shared__ float s_b5;

    const int tid  = threadIdx.x;
    const int wave = tid >> 5;
    const int lane = tid & 31;
    const int bid  = blockIdx.x;
    const int b    = bid >> 9;          // / NN
    const int i    = bid & (NN - 1);

    if (tid < CC) s_xi[tid] = x[((size_t)(b * NN + i)) * CC + tid];
    if (tid == 0) s_b5 = b5[0];
    __syncthreads();

    const int hi = lane >> 4;
    const int nl = lane & 15;
    const float cb5 = s_b5;
    _Float16* bufA = &s_buf[wave][0][0];
    _Float16* bufB = &s_buf[wave][1][0];

#pragma unroll 1
    for (int t = 0; t < 8; ++t) {
        const int j0 = ((wave << 3) + t) << 4;
        const float* xj = x + ((size_t)(b * NN + j0 + nl)) * CC;

        // ---- Layer 1: build A = |x_i - x_j| fragments directly (K=128)
        v16h a1[KT1];
#pragma unroll
        for (int kf = 0; kf < KT1; ++kf) {
            const int c0 = (kf << 5) + (hi << 3);
            v4f ja = *(const v4f*)(xj + c0);
            v4f jb = *(const v4f*)(xj + c0 + 4);
            v4f jc = *(const v4f*)(xj + c0 + 16);
            v4f jd = *(const v4f*)(xj + c0 + 20);
            v4f ia = *(const v4f*)(s_xi + c0);
            v4f ib = *(const v4f*)(s_xi + c0 + 4);
            v4f ic = *(const v4f*)(s_xi + c0 + 16);
            v4f id = *(const v4f*)(s_xi + c0 + 20);
            v16h a;
#pragma unroll
            for (int e = 0; e < 4; ++e) {
                a[e]      = (_Float16)fabsf(ja[e] - ia[e]);
                a[e + 4]  = (_Float16)fabsf(jb[e] - ib[e]);
                a[e + 8]  = (_Float16)fabsf(jc[e] - ic[e]);
                a[e + 12] = (_Float16)fabsf(jd[e] - id[e]);
            }
            a1[kf] = a;
        }
        gemm_store<KT1, NT1>(a1, wf + OFF_W1, b1, bufA, HH, lane);
        asm volatile("s_wait_dscnt 0" ::: "memory");  // order LDS store -> load (same wave)

        run_layer<KT2, NT2>(bufA, HH, wf + OFF_W2, b2, bufB, HH, lane);
        asm volatile("s_wait_dscnt 0" ::: "memory");
        run_layer<KT3, NT3>(bufB, HH, wf + OFF_W3, b3, bufA, NFE, lane);
        asm volatile("s_wait_dscnt 0" ::: "memory");
        run_layer<KT4, NT4>(bufA, NFE, wf + OFF_W4, b4, bufB, NFE, lane);
        asm volatile("s_wait_dscnt 0" ::: "memory");

        // ---- Layer 5: 96 -> 1 as 3 WMMAs against broadcast w5 fragments.
        // Every column of D holds the logit for its row; column-0 lanes
        // (nl == 0, i.e. lanes 0 and 16) publish rows 8*hi .. 8*hi+7.
        {
            v16h a5[KT5];
#pragma unroll
            for (int kf = 0; kf < KT5; ++kf) a5[kf] = ld_afrag(bufB, lane, kf, NFE);
            v8f acc = {};
#pragma unroll
            for (int kf = 0; kf < KT5; ++kf) {
                v16h bfr = ld_bfrag(wf + OFF_W5, kf, lane);
                acc = __builtin_amdgcn_wmma_f32_16x16x32_f16(
                    false, a5[kf], false, bfr, (short)0, acc, false, false);
            }
            if (nl == 0) {
                float* q = &s_wrow[j0 + (hi << 3)];
#pragma unroll
                for (int v = 0; v < 8; ++v) q[v] = acc[v] + cb5;  // 2x ds_store_b128
            }
        }
    }
    __syncthreads();

    // ---- masked softmax over j (512 values, 128 threads, 4 each)
    float vloc[4];
    float lmax = -3.0e38f;
#pragma unroll
    for (int k = 0; k < 4; ++k) {
        int j = tid + (k << 7);
        float v = s_wrow[j];
        if (j == i) v -= 1.0e8f;  // mask self-edge
        vloc[k] = v;
        lmax = fmaxf(lmax, v);
    }
    s_red[tid] = lmax;
    __syncthreads();
    for (int ss = 64; ss > 0; ss >>= 1) {
        if (tid < ss) s_red[tid] = fmaxf(s_red[tid], s_red[tid + ss]);
        __syncthreads();
    }
    const float m = s_red[0];
    __syncthreads();
    float lsum = 0.f;
#pragma unroll
    for (int k = 0; k < 4; ++k) {
        vloc[k] = __expf(vloc[k] - m);
        lsum += vloc[k];
    }
    s_red[tid] = lsum;
    __syncthreads();
    for (int ss = 64; ss > 0; ss >>= 1) {
        if (tid < ss) s_red[tid] += s_red[tid + ss];
        __syncthreads();
    }
    const float inv = 1.f / s_red[0];
#pragma unroll
    for (int k = 0; k < 4; ++k) {
        int j = tid + (k << 7);
        size_t o = (((size_t)bid) * NN + j) * 2;   // bid == b*NN + i
        out[o]     = (j == i) ? 1.0f : 0.0f;       // W_id channel
        out[o + 1] = vloc[k] * inv;                // softmax channel
    }
}

// ---------------------------------------------------------------------------
extern "C" void kernel_launch(void* const* d_in, const int* in_sizes, int n_in,
                              void* d_out, int out_size, void* d_ws, size_t ws_size,
                              hipStream_t stream) {
    const float* x  = (const float*)d_in[0];
    // d_in[1] = W_id (identity mask) -- reconstructed analytically, unused
    const float* w1 = (const float*)d_in[2];
    const float* b1 = (const float*)d_in[3];
    const float* w2 = (const float*)d_in[4];
    const float* b2 = (const float*)d_in[5];
    const float* w3 = (const float*)d_in[6];
    const float* b3 = (const float*)d_in[7];
    const float* w4 = (const float*)d_in[8];
    const float* b4 = (const float*)d_in[9];
    const float* w5 = (const float*)d_in[10];
    const float* b5 = (const float*)d_in[11];
    _Float16* wfr = (_Float16*)d_ws;   // needs ~181 KB of workspace
    float* out = (float*)d_out;

    prep_wfrag<<<(KT1 * NT1 * 512 + 255) / 256, 256, 0, stream>>>(w1, wfr + OFF_W1, CC, NT1, KT1 * NT1 * 512);
    prep_wfrag<<<(KT2 * NT2 * 512 + 255) / 256, 256, 0, stream>>>(w2, wfr + OFF_W2, HH, NT2, KT2 * NT2 * 512);
    prep_wfrag<<<(KT3 * NT3 * 512 + 255) / 256, 256, 0, stream>>>(w3, wfr + OFF_W3, HH, NT3, KT3 * NT3 * 512);
    prep_wfrag<<<(KT4 * NT4 * 512 + 255) / 256, 256, 0, stream>>>(w4, wfr + OFF_W4, NFE, NT4, KT4 * NT4 * 512);
    prep_w5frag<<<(KT5 * 512 + 255) / 256, 256, 0, stream>>>(w5, wfr + OFF_W5, KT5 * 512);

    edge_mlp_softmax<<<BB * NN, 128, 0, stream>>>(x, b1, b2, b3, b4, b5, wfr, out);
}
